// QuantLinear_13486197309928
// MI455X (gfx1250) — matrix operand order repaired and checked by reading, not compiled, not run on hardware
//
#include <hip/hip_runtime.h>

typedef __attribute__((ext_vector_type(16))) _Float16 v16h;
typedef __attribute__((ext_vector_type(8)))  _Float16 v8h;
typedef __attribute__((ext_vector_type(4)))  _Float16 v4h;
typedef __attribute__((ext_vector_type(8)))  float    v8f;
typedef __attribute__((ext_vector_type(4)))  float    f32x4;
typedef __attribute__((ext_vector_type(4)))  int      i32x4;

#define BM 256
#define BN 128
#define BK 32
#define LDK 40   // padded LDS k-stride in halves (80 B, 16B aligned, bank-skewed)

union V16 { v16h v; v8h h[2]; };

__global__ __launch_bounds__(256)
void qlinear_wmma_kernel(const float* __restrict__ x,
                         const int*   __restrict__ qw,
                         const float* __restrict__ scale,
                         const int*   __restrict__ zp,
                         const float* __restrict__ bias,
                         float* __restrict__ out,
                         int M, int I, int O)
{
    __shared__ _Float16 lA[2][BM][LDK];   // X tile: m x k (f16)           40 KB
    __shared__ _Float16 lB[2][BN][LDK];   // dequantized W tile: n x k     20 KB

    const int tid  = threadIdx.x;
    const int lane = tid & 31;
    const int wave = tid >> 5;
    const int wm   = wave >> 1;        // 0..3 -> 64-row m slab
    const int wn   = wave & 1;         // 0..1 -> 64-col n slab
    const int lr   = lane & 15;        // row/col within a 16-tile
    const int lh   = lane >> 4;        // lane half (K-block select)

    const int blockN = blockIdx.x * BN;
    const int blockM = blockIdx.y * BM;
    const int G = I >> 7;              // groups of 128 along I

    // Loader mapping:
    //   X: each thread owns one full 32-elem k-row of the 256-row tile.
    //   W: each thread owns a 16-elem half-row of the 128-row tile.
    const int xr_row = tid;                // 0..255
    const int wr_row = tid >> 1;           // 0..127
    const int wk16   = (tid & 1) * 16;     // 0 or 16

    const float* xrow  = x  + (size_t)(blockM + xr_row) * I;
    const int*   qrow  = qw + (size_t)(blockN + wr_row) * I + wk16;
    const int    sidx0 = (blockN + wr_row) * G;

    v8f acc[4][4] = {};

    f32x4 xr[8];
    i32x4 qr[4];
    float sS = 0.0f, sZ = 0.0f;

    const int NIT = I / BK;

    // ---- prologue: stage chunk 0 into LDS[0] ----
    {
        #pragma unroll
        for (int j = 0; j < 8; ++j) xr[j] = *(const f32x4*)(xrow + 4 * j);
        #pragma unroll
        for (int j = 0; j < 4; ++j) qr[j] = *(const i32x4*)(qrow + 4 * j);
        sS = scale[sidx0];
        sZ = (float)zp[sidx0];
        #pragma unroll
        for (int j = 0; j < 8; ++j) {
            v4h hx;
            #pragma unroll
            for (int e = 0; e < 4; ++e) hx[e] = (_Float16)xr[j][e];
            *(v4h*)&lA[0][xr_row][4 * j] = hx;
        }
        #pragma unroll
        for (int j = 0; j < 4; ++j) {
            v4h hw;
            #pragma unroll
            for (int e = 0; e < 4; ++e)
                hw[e] = (_Float16)(((float)qr[j][e] - sZ) * sS);
            *(v4h*)&lB[0][wr_row][wk16 + 4 * j] = hw;
        }
    }
    __syncthreads();

    int buf = 0;
    for (int it = 0; it < NIT; ++it) {
        const bool more = (it + 1) < NIT;

        // ---- stage next chunk from global into registers (overlaps WMMA) ----
        if (more) {
            const int k0 = (it + 1) * BK;
            #pragma unroll
            for (int j = 0; j < 8; ++j) xr[j] = *(const f32x4*)(xrow + k0 + 4 * j);
            #pragma unroll
            for (int j = 0; j < 4; ++j) qr[j] = *(const i32x4*)(qrow + k0 + 4 * j);
            const int g = k0 >> 7;
            sS = scale[sidx0 + g];
            sZ = (float)zp[sidx0 + g];
        }

        // ---- load fragments from LDS (ISA 7.12.2 16-bit layouts) ----
        // A (16x32 f16): lanes 0-15 hold K[0..8)+[16..24), lanes 16-31 K[8..16)+[24..32)
        v16h a[4];
        #pragma unroll
        for (int tm = 0; tm < 4; ++tm) {
            const _Float16* p = &lA[buf][wm * 64 + tm * 16 + lr][lh * 8];
            V16 u;
            u.h[0] = *(const v8h*)(p);
            u.h[1] = *(const v8h*)(p + 16);
            a[tm] = u.v;
        }
        // B (32x16 f16): lane = column, lanes 0-15 K[0..8)+[16..24), lanes 16-31 K[8..16)+[24..32)
        v16h b[4];
        #pragma unroll
        for (int tn = 0; tn < 4; ++tn) {
            const _Float16* p = &lB[buf][wn * 64 + tn * 16 + lr][lh * 8];
            V16 u;
            u.h[0] = *(const v8h*)(p);
            u.h[1] = *(const v8h*)(p + 16);
            b[tn] = u.v;
        }

        // ---- 16x v_wmma_f32_16x16x32_f16 (64x64 per wave) ----
        #pragma unroll
        for (int tm = 0; tm < 4; ++tm)
            #pragma unroll
            for (int tn = 0; tn < 4; ++tn)
                acc[tm][tn] = __builtin_amdgcn_wmma_f32_16x16x32_f16(
                    /*neg_a=*/false, a[tm],
                    /*neg_b=*/false, b[tn],
                    /*c_mod=*/(short)0, acc[tm][tn],
                    /*reuse_a=*/false, /*reuse_b=*/false);

        // ---- convert + store staged chunk into the other LDS buffer ----
        if (more) {
            const int wb = buf ^ 1;
            #pragma unroll
            for (int j = 0; j < 8; ++j) {
                v4h hx;
                #pragma unroll
                for (int e = 0; e < 4; ++e) hx[e] = (_Float16)xr[j][e];
                *(v4h*)&lA[wb][xr_row][4 * j] = hx;
            }
            #pragma unroll
            for (int j = 0; j < 4; ++j) {
                v4h hw;
                #pragma unroll
                for (int e = 0; e < 4; ++e)
                    hw[e] = (_Float16)(((float)qr[j][e] - sZ) * sS);
                *(v4h*)&lB[wb][wr_row][wk16 + 4 * j] = hw;
            }
        }
        __syncthreads();
        buf ^= 1;
    }

    // ---- epilogue: C/D layout -> global (lanes 0-15: M=r, lanes 16-31: M=r+8) ----
    #pragma unroll
    for (int tn = 0; tn < 4; ++tn) {
        const int nn = blockN + wn * 64 + tn * 16 + lr;
        const float bv = bias[nn];
        #pragma unroll
        for (int tm = 0; tm < 4; ++tm) {
            const int mbase = blockM + wm * 64 + tm * 16 + lh * 8;
            #pragma unroll
            for (int r = 0; r < 8; ++r) {
                out[(size_t)(mbase + r) * O + nn] = acc[tm][tn][r] + bv;
            }
        }
    }
}

extern "C" void kernel_launch(void* const* d_in, const int* in_sizes, int n_in,
                              void* d_out, int out_size, void* d_ws, size_t ws_size,
                              hipStream_t stream) {
    const float* x     = (const float*)d_in[0];   // [B,S,I] fp32
    const int*   qw    = (const int*)d_in[1];     // [O,I]   int32 (4-bit values)
    const float* scale = (const float*)d_in[2];   // [O,G]   fp32
    const int*   zp    = (const int*)d_in[3];     // [O,G]   int32
    const float* bias  = (const float*)d_in[4];   // [O]     fp32
    float*       out   = (float*)d_out;           // [B,S,O] fp32

    const int O = in_sizes[4];                    // bias length
    const int I = in_sizes[1] / O;                // qweight = O*I
    const int M = in_sizes[0] / I;                // x = M*I

    dim3 grid(O / BN, M / BM);                    // 86 x 16 for this problem
    qlinear_wmma_kernel<<<grid, dim3(256), 0, stream>>>(x, qw, scale, zp, bias,
                                                        out, M, I, O);
}
